// TransformerEncoder_15857019257396
// MI455X (gfx1250) — matrix operand order repaired and checked by reading, not compiled, not run on hardware
//
#include <hip/hip_runtime.h>

// ---------------------------------------------------------------------------
// Types / constants
// ---------------------------------------------------------------------------
typedef __bf16 bf16_t;
typedef __attribute__((ext_vector_type(16))) __bf16        v16bf;
typedef __attribute__((ext_vector_type(8)))  float         v8f;
typedef __attribute__((ext_vector_type(8)))  unsigned int  v8u;

#define S_SEQ  4096
#define DMODEL 1024
#define NHEAD  16
#define DHEAD  64
#define FFDIM  4096

#if __has_builtin(__builtin_amdgcn_global_load_async_to_lds_b32)
#define HAS_ASYNC_LDS 1
#else
#define HAS_ASYNC_LDS 0
#endif

static __device__ __forceinline__ bf16_t f2bf(float f) {
  unsigned u = __builtin_bit_cast(unsigned, f);
  u += 0x7FFFu + ((u >> 16) & 1u);               // round-to-nearest-even
  unsigned short h = (unsigned short)(u >> 16);
  return __builtin_bit_cast(bf16_t, h);
}

static __device__ __forceinline__ v8f vzero8() {
  v8f v;
#pragma unroll
  for (int i = 0; i < 8; ++i) v[i] = 0.0f;
  return v;
}

// global -> LDS async copy of 4 bytes (2 bf16); no VGPR round-trip.
static __device__ __forceinline__ void async_copy_b32(const bf16_t* g, bf16_t* l) {
#if HAS_ASYNC_LDS
  __builtin_amdgcn_global_load_async_to_lds_b32(
      (__attribute__((address_space(1))) int*)(g),
      (__attribute__((address_space(3))) int*)(l), 0, 0);
#else
  (void)g; (void)l;
#endif
}

static __device__ __forceinline__ void async_wait_all() {
#if HAS_ASYNC_LDS
#if __has_builtin(__builtin_amdgcn_s_wait_asynccnt)
  __builtin_amdgcn_s_wait_asynccnt(0);
#else
  asm volatile("s_wait_asynccnt 0" ::: "memory");
#endif
#endif
}

// ---------------------------------------------------------------------------
// fp32 -> bf16 conversion (weights)
// ---------------------------------------------------------------------------
__global__ void f32_to_bf16(const float* __restrict__ s, bf16_t* __restrict__ d, int n) {
  int i = blockIdx.x * blockDim.x + threadIdx.x;
  const int stride = gridDim.x * blockDim.x;
  for (; i < n; i += stride) d[i] = f2bf(s[i]);
}

// ---------------------------------------------------------------------------
// LayerNorm: fp32 [row, DMODEL] -> bf16
// ---------------------------------------------------------------------------
__global__ __launch_bounds__(256) void layernorm_bf16(
    const float* __restrict__ x, const float* __restrict__ sc,
    const float* __restrict__ bi, bf16_t* __restrict__ y) {
  __shared__ float red[256];
  const int row = blockIdx.x;
  const int tid = threadIdx.x;
  const float* xr = x + (size_t)row * DMODEL;

  float loc[4];
  float sum = 0.0f;
#pragma unroll
  for (int j = 0; j < 4; ++j) { loc[j] = xr[tid + j * 256]; sum += loc[j]; }
  red[tid] = sum; __syncthreads();
  for (int s = 128; s > 0; s >>= 1) { if (tid < s) red[tid] += red[tid + s]; __syncthreads(); }
  const float mu = red[0] * (1.0f / DMODEL);
  __syncthreads();

  float sq = 0.0f;
#pragma unroll
  for (int j = 0; j < 4; ++j) { const float d = loc[j] - mu; sq += d * d; }
  red[tid] = sq; __syncthreads();
  for (int s = 128; s > 0; s >>= 1) { if (tid < s) red[tid] += red[tid + s]; __syncthreads(); }
  const float rs = rsqrtf(red[0] * (1.0f / DMODEL) + 1e-6f);

#pragma unroll
  for (int j = 0; j < 4; ++j) {
    const int c = tid + j * 256;
    y[(size_t)row * DMODEL + c] = f2bf((loc[j] - mu) * rs * sc[c] + bi[c]);
  }
}

// ---------------------------------------------------------------------------
// Tiled bf16 WMMA GEMM: C[M,N] = A[M,K] * B[K,N] (+ epilogue)
//   EPI 0: outB = acc + bias                 (K/V projection)
//   EPI 1: outB = (acc + bias) * scale       (Q projection, folds 1/sqrt(DH))
//   EPI 2: outF = acc + bias + resid         (O-proj / FFN2 + residual)
//   EPI 3: outB = gelu(acc + bias)           (FFN1)
// Block: 256 threads = 8 waves (4 along M x 2 along N), tile 128x128, BK=32.
// Double-buffered LDS; A tile staged via GLOBAL_LOAD_ASYNC_TO_LDS (ASYNCcnt)
// so the next tile's DMA overlaps the current tile's WMMAs.
// One barrier per K-step.
// ---------------------------------------------------------------------------
template <int EPI>
__global__ __launch_bounds__(256) void gemm_bf16_wmma(
    const bf16_t* __restrict__ A, const bf16_t* __restrict__ B,
    const float* __restrict__ bias, const float* __restrict__ resid,
    float scale, float* __restrict__ outF, bf16_t* __restrict__ outB,
    int M, int N, int K) {
  constexpr int BM = 128, BN = 128, BK = 32;
  constexpr int LDA = BK + 2;   // 34 elems = 17 dwords -> conflict-free
  constexpr int LDB = BK + 2;
  __shared__ bf16_t As[2][BM * LDA];     // A tile, row-major (double-buffered)
  __shared__ bf16_t Bs[2][BN * LDB];     // B tile, transposed: Bs[n][k]

  const int tid  = threadIdx.x;
  const int lane = tid & 31;
  const int wid  = tid >> 5;
  const int half = lane >> 4;
  const int l16  = lane & 15;
  const int wm   = wid & 3;              // wave row (4 along M)
  const int wn   = wid >> 2;             // wave col (2 along N)

  const int mBlk = blockIdx.y * BM;
  const int nBlk = blockIdx.x * BN;

  v8f acc[2][4];
#pragma unroll
  for (int i = 0; i < 2; ++i)
#pragma unroll
    for (int j = 0; j < 4; ++j) acc[i][j] = vzero8();

  const int aRow = tid >> 1, aSeg = (tid & 1) * 16;  // 128 rows x 32 cols
  const int bRow = tid >> 3, bSeg = (tid & 7) * 16;  // 32 rows x 128 cols

  // stage tile at k0 into buffer buf
  auto stage = [&](int k0, int buf) {
    {  // A tile: async DMA straight into LDS when supported
      const bf16_t* g = A + (size_t)(mBlk + aRow) * K + k0 + aSeg;
      bf16_t* s = &As[buf][aRow * LDA + aSeg];
#if HAS_ASYNC_LDS
#pragma unroll
      for (int i = 0; i < 8; ++i) async_copy_b32(g + 2 * i, s + 2 * i);
#else
      const unsigned* gu = (const unsigned*)g;
      unsigned* su = (unsigned*)s;
#pragma unroll
      for (int i = 0; i < 8; ++i) su[i] = gu[i];
#endif
    }
    {  // B tile: transpose through VGPRs into Bs[n][k]
      const unsigned* g = (const unsigned*)(B + (size_t)(k0 + bRow) * N + nBlk + bSeg);
#pragma unroll
      for (int i = 0; i < 8; ++i) {
        const unsigned p = g[i];
        const int n = bSeg + 2 * i;
        Bs[buf][(n + 0) * LDB + bRow] = __builtin_bit_cast(bf16_t, (unsigned short)(p & 0xFFFFu));
        Bs[buf][(n + 1) * LDB + bRow] = __builtin_bit_cast(bf16_t, (unsigned short)(p >> 16));
      }
    }
  };

  stage(0, 0);
  async_wait_all();
  __syncthreads();

  const int nSteps = K / BK;
  for (int t = 0; t < nSteps; ++t) {
    const int cur = t & 1;
    if (t + 1 < nSteps) stage((t + 1) * BK, cur ^ 1);

    v16bf afrag[2], bfrag[4];
#pragma unroll
    for (int mi = 0; mi < 2; ++mi) {     // A 16x32: row=lane&15, K per ISA table
      const int mrow = wm * 32 + mi * 16 + l16;
      v8u au;
#pragma unroll
      for (int i = 0; i < 8; ++i) {
        const int kk = (i < 4 ? 2 * i : 2 * i + 8) + half * 8;
        au[i] = *(const unsigned*)&As[cur][mrow * LDA + kk];
      }
      afrag[mi] = __builtin_bit_cast(v16bf, au);
    }
#pragma unroll
    for (int ni = 0; ni < 4; ++ni) {     // B 32x16: col=lane&15, K per ISA table
      const int ncol = wn * 64 + ni * 16 + l16;
      v8u bu;
#pragma unroll
      for (int i = 0; i < 8; ++i) {
        const int kk = (i < 4 ? 2 * i : 2 * i + 8) + half * 8;
        bu[i] = *(const unsigned*)&Bs[cur][ncol * LDB + kk];
      }
      bfrag[ni] = __builtin_bit_cast(v16bf, bu);
    }
#pragma unroll
    for (int mi = 0; mi < 2; ++mi)
#pragma unroll
      for (int ni = 0; ni < 4; ++ni)
        acc[mi][ni] = __builtin_amdgcn_wmma_f32_16x16x32_bf16(
            false, afrag[mi], false, bfrag[ni], (short)0, acc[mi][ni], false, false);

    async_wait_all();     // next tile's DMA must have landed in LDS
    __syncthreads();      // one barrier per K-step
  }

  // Epilogue (C/D layout: row = vgpr + 8*half, col = lane&15)
#pragma unroll
  for (int mi = 0; mi < 2; ++mi) {
#pragma unroll
    for (int ni = 0; ni < 4; ++ni) {
      const int gn = nBlk + wn * 64 + ni * 16 + l16;
      const float bb = bias[gn];
#pragma unroll
      for (int r = 0; r < 8; ++r) {
        const int gm = mBlk + wm * 32 + mi * 16 + half * 8 + r;
        float v = acc[mi][ni][r] + bb;
        const size_t idx = (size_t)gm * N + gn;
        if (EPI == 0) {
          outB[idx] = f2bf(v);
        } else if (EPI == 1) {
          outB[idx] = f2bf(v * scale);
        } else if (EPI == 2) {
          outF[idx] = v + resid[idx];
        } else {
          v = 0.5f * v * (1.0f + erff(v * 0.70710678118654752f));  // exact gelu
          outB[idx] = f2bf(v);
        }
      }
    }
  }
}

// ---------------------------------------------------------------------------
// Flash attention: one block per (64-query tile, head). 128 threads = 4 waves.
// Wave w owns query rows w*16..w*16+15 so softmax stats stay wave-local.
// q already carries the 1/sqrt(DH) scale (folded into Q projection epilogue).
// q/k/v/ctx layout: [S, H*DH] bf16.
// ---------------------------------------------------------------------------
__global__ __launch_bounds__(128) void flash_attn(
    const bf16_t* __restrict__ q, const bf16_t* __restrict__ k,
    const bf16_t* __restrict__ v, bf16_t* __restrict__ ctx) {
  constexpr int LDT = DHEAD + 2;   // 66 elems = 33 dwords, conflict-free
  __shared__ bf16_t Qs[64 * LDT];  // Q row-major (A source)
  __shared__ bf16_t Ks[64 * LDT];  // K row-major == B^T layout for Q*K^T
  __shared__ bf16_t Vs[64 * LDT];  // V transposed: Vs[d][s] (B^T layout for P*V)
  __shared__ bf16_t Ps[64 * LDT];  // probs, re-shaped C-layout -> A-layout

  const int h   = blockIdx.y;
  const int q0  = blockIdx.x * 64;
  const int tid = threadIdx.x;
  const int lane = tid & 31;
  const int wid  = tid >> 5;
  const int half = lane >> 4;
  const int l16  = lane & 15;

  const int lRow = tid >> 1, lSeg = (tid & 1) * 32;   // 64 rows x 64 cols staging

  {   // load Q tile once
    const unsigned* g = (const unsigned*)(q + (size_t)(q0 + lRow) * DMODEL + h * DHEAD + lSeg);
    unsigned* s = (unsigned*)&Qs[lRow * LDT + lSeg];
#pragma unroll
    for (int i = 0; i < 16; ++i) s[i] = g[i];
  }

  v8f o[4];
#pragma unroll
  for (int j = 0; j < 4; ++j) o[j] = vzero8();
  float mrun[8], lrun[8];
#pragma unroll
  for (int r = 0; r < 8; ++r) { mrun[r] = -1e30f; lrun[r] = 0.0f; }

  for (int t0 = 0; t0 < S_SEQ; t0 += 64) {
    __syncthreads();
    {   // K tile, row-major
      const unsigned* g = (const unsigned*)(k + (size_t)(t0 + lRow) * DMODEL + h * DHEAD + lSeg);
      unsigned* s = (unsigned*)&Ks[lRow * LDT + lSeg];
#pragma unroll
      for (int i = 0; i < 16; ++i) s[i] = g[i];
    }
    {   // V tile, transposed into Vs[d][s]
      const unsigned* g = (const unsigned*)(v + (size_t)(t0 + lRow) * DMODEL + h * DHEAD + lSeg);
#pragma unroll
      for (int i = 0; i < 16; ++i) {
        const unsigned p = g[i];
        const int d = lSeg + 2 * i;
        Vs[(d + 0) * LDT + lRow] = __builtin_bit_cast(bf16_t, (unsigned short)(p & 0xFFFFu));
        Vs[(d + 1) * LDT + lRow] = __builtin_bit_cast(bf16_t, (unsigned short)(p >> 16));
      }
    }
    if (t0 + 64 < S_SEQ) {   // speculative prefetch of the next K/V tiles
      __builtin_prefetch(k + (size_t)(t0 + 64 + lRow) * DMODEL + h * DHEAD + lSeg, 0, 0);
      __builtin_prefetch(v + (size_t)(t0 + 64 + lRow) * DMODEL + h * DHEAD + lSeg, 0, 0);
    }
    __syncthreads();

    // ---- S = Q * K^T  (16x64 per wave, 8 WMMAs) ----
    v8f sc[4];
#pragma unroll
    for (int j = 0; j < 4; ++j) sc[j] = vzero8();
#pragma unroll
    for (int kk = 0; kk < 2; ++kk) {
      v16bf aq;
      {
        const int mrow = wid * 16 + l16;
        v8u au;
#pragma unroll
        for (int i = 0; i < 8; ++i) {
          const int kc = kk * 32 + (i < 4 ? 2 * i : 2 * i + 8) + half * 8;
          au[i] = *(const unsigned*)&Qs[mrow * LDT + kc];
        }
        aq = __builtin_bit_cast(v16bf, au);
      }
#pragma unroll
      for (int j = 0; j < 4; ++j) {
        const int ncol = j * 16 + l16;
        v8u bu;
#pragma unroll
        for (int i = 0; i < 8; ++i) {
          const int kc = kk * 32 + (i < 4 ? 2 * i : 2 * i + 8) + half * 8;
          bu[i] = *(const unsigned*)&Ks[ncol * LDT + kc];
        }
        const v16bf bk_ = __builtin_bit_cast(v16bf, bu);
        sc[j] = __builtin_amdgcn_wmma_f32_16x16x32_bf16(
            false, aq, false, bk_, (short)0, sc[j], false, false);
      }
    }

    // ---- online softmax; row r lives across the lane-group's 16 lanes ----
#pragma unroll
    for (int r = 0; r < 8; ++r) {
      float rmax = sc[0][r];
#pragma unroll
      for (int j = 1; j < 4; ++j) rmax = fmaxf(rmax, sc[j][r]);
#pragma unroll
      for (int off = 8; off >= 1; off >>= 1) rmax = fmaxf(rmax, __shfl_xor(rmax, off, 32));
      const float mnew  = fmaxf(mrun[r], rmax);
      const float alpha = __expf(mrun[r] - mnew);
      float rsum = 0.0f;
#pragma unroll
      for (int j = 0; j < 4; ++j) {
        const float p = __expf(sc[j][r] - mnew);
        sc[j][r] = p;
        rsum += p;
      }
#pragma unroll
      for (int off = 8; off >= 1; off >>= 1) rsum += __shfl_xor(rsum, off, 32);
      lrun[r] = lrun[r] * alpha + rsum;
      mrun[r] = mnew;
#pragma unroll
      for (int j = 0; j < 4; ++j) o[j][r] = o[j][r] * alpha;
    }

    // ---- stash P (bf16), wave-private rows, then O += P * V (8 WMMAs) ----
#pragma unroll
    for (int j = 0; j < 4; ++j)
#pragma unroll
      for (int r = 0; r < 8; ++r)
        Ps[(wid * 16 + half * 8 + r) * LDT + j * 16 + l16] = f2bf(sc[j][r]);

#pragma unroll
    for (int kk = 0; kk < 2; ++kk) {
      v16bf ap;
      {
        const int mrow = wid * 16 + l16;
        v8u au;
#pragma unroll
        for (int i = 0; i < 8; ++i) {
          const int kc = kk * 32 + (i < 4 ? 2 * i : 2 * i + 8) + half * 8;
          au[i] = *(const unsigned*)&Ps[mrow * LDT + kc];
        }
        ap = __builtin_bit_cast(v16bf, au);
      }
#pragma unroll
      for (int j = 0; j < 4; ++j) {
        const int ncol = j * 16 + l16;
        v8u bu;
#pragma unroll
        for (int i = 0; i < 8; ++i) {
          const int kc = kk * 32 + (i < 4 ? 2 * i : 2 * i + 8) + half * 8;
          bu[i] = *(const unsigned*)&Vs[ncol * LDT + kc];
        }
        const v16bf bv_ = __builtin_bit_cast(v16bf, bu);
        o[j] = __builtin_amdgcn_wmma_f32_16x16x32_bf16(
            false, ap, false, bv_, (short)0, o[j], false, false);
      }
    }
  }

  // normalize by l and emit ctx (bf16)
#pragma unroll
  for (int j = 0; j < 4; ++j) {
#pragma unroll
    for (int r = 0; r < 8; ++r) {
      const int row = q0 + wid * 16 + half * 8 + r;
      const int d   = j * 16 + l16;
      ctx[(size_t)row * DMODEL + h * DHEAD + d] = f2bf(o[j][r] / lrun[r]);
    }
  }
}

// ---------------------------------------------------------------------------
// Launch
// ---------------------------------------------------------------------------
extern "C" void kernel_launch(void* const* d_in, const int* in_sizes, int n_in,
                              void* d_out, int out_size, void* d_ws, size_t ws_size,
                              hipStream_t stream) {
  const float* x    = (const float*)d_in[0];
  const float* Wq   = (const float*)d_in[1];
  const float* bq   = (const float*)d_in[2];
  const float* Wk   = (const float*)d_in[3];
  const float* bk   = (const float*)d_in[4];
  const float* Wv   = (const float*)d_in[5];
  const float* bv   = (const float*)d_in[6];
  const float* Wo   = (const float*)d_in[7];
  const float* bo   = (const float*)d_in[8];
  const float* ln1s = (const float*)d_in[9];
  const float* ln1b = (const float*)d_in[10];
  const float* ln2s = (const float*)d_in[11];
  const float* ln2b = (const float*)d_in[12];
  const float* W1   = (const float*)d_in[13];
  const float* b1   = (const float*)d_in[14];
  const float* W2   = (const float*)d_in[15];
  const float* b2   = (const float*)d_in[16];
  float* out = (float*)d_out;

  char* ws = (char*)d_ws;
  size_t off = 0;
  auto alloc = [&](size_t bytes) -> char* {
    char* p = ws + off;
    off += (bytes + 255) & ~(size_t)255;
    return p;
  };
  bf16_t* Wq_b = (bf16_t*)alloc((size_t)DMODEL * DMODEL * 2);
  bf16_t* Wk_b = (bf16_t*)alloc((size_t)DMODEL * DMODEL * 2);
  bf16_t* Wv_b = (bf16_t*)alloc((size_t)DMODEL * DMODEL * 2);
  bf16_t* Wo_b = (bf16_t*)alloc((size_t)DMODEL * DMODEL * 2);
  bf16_t* W1_b = (bf16_t*)alloc((size_t)DMODEL * FFDIM * 2);
  bf16_t* W2_b = (bf16_t*)alloc((size_t)FFDIM * DMODEL * 2);
  bf16_t* y1   = (bf16_t*)alloc((size_t)S_SEQ * DMODEL * 2);
  bf16_t* qb   = (bf16_t*)alloc((size_t)S_SEQ * DMODEL * 2);
  bf16_t* kb   = (bf16_t*)alloc((size_t)S_SEQ * DMODEL * 2);
  bf16_t* vb   = (bf16_t*)alloc((size_t)S_SEQ * DMODEL * 2);
  bf16_t* ctxb = (bf16_t*)alloc((size_t)S_SEQ * DMODEL * 2);
  float*  x2   = (float*) alloc((size_t)S_SEQ * DMODEL * 4);
  bf16_t* y2   = (bf16_t*)alloc((size_t)S_SEQ * DMODEL * 2);
  bf16_t* h1   = (bf16_t*)alloc((size_t)S_SEQ * FFDIM * 2);

  // weights -> bf16
  f32_to_bf16<<<512, 256, 0, stream>>>(Wq, Wq_b, DMODEL * DMODEL);
  f32_to_bf16<<<512, 256, 0, stream>>>(Wk, Wk_b, DMODEL * DMODEL);
  f32_to_bf16<<<512, 256, 0, stream>>>(Wv, Wv_b, DMODEL * DMODEL);
  f32_to_bf16<<<512, 256, 0, stream>>>(Wo, Wo_b, DMODEL * DMODEL);
  f32_to_bf16<<<512, 256, 0, stream>>>(W1, W1_b, DMODEL * FFDIM);
  f32_to_bf16<<<512, 256, 0, stream>>>(W2, W2_b, FFDIM * DMODEL);

  // LN1
  layernorm_bf16<<<S_SEQ, 256, 0, stream>>>(x, ln1s, ln1b, y1);

  // QKV projections (scale = 1/sqrt(DH) folded into Q)
  const dim3 gD(DMODEL / 128, S_SEQ / 128);
  gemm_bf16_wmma<1><<<gD, 256, 0, stream>>>(y1, Wq_b, bq, nullptr, 0.125f,
                                            nullptr, qb, S_SEQ, DMODEL, DMODEL);
  gemm_bf16_wmma<0><<<gD, 256, 0, stream>>>(y1, Wk_b, bk, nullptr, 1.0f,
                                            nullptr, kb, S_SEQ, DMODEL, DMODEL);
  gemm_bf16_wmma<0><<<gD, 256, 0, stream>>>(y1, Wv_b, bv, nullptr, 1.0f,
                                            nullptr, vb, S_SEQ, DMODEL, DMODEL);

  // attention
  flash_attn<<<dim3(S_SEQ / 64, NHEAD), 128, 0, stream>>>(qb, kb, vb, ctxb);

  // O-projection + residual -> x2 (fp32)
  gemm_bf16_wmma<2><<<gD, 256, 0, stream>>>(ctxb, Wo_b, bo, x, 1.0f,
                                            x2, nullptr, S_SEQ, DMODEL, DMODEL);

  // LN2
  layernorm_bf16<<<S_SEQ, 256, 0, stream>>>(x2, ln2s, ln2b, y2);

  // FFN1 (+gelu) and FFN2 (+residual)
  const dim3 gF(FFDIM / 128, S_SEQ / 128);
  gemm_bf16_wmma<3><<<gF, 256, 0, stream>>>(y2, W1_b, b1, nullptr, 1.0f,
                                            nullptr, h1, S_SEQ, FFDIM, DMODEL);
  gemm_bf16_wmma<2><<<gD, 256, 0, stream>>>(h1, W2_b, b2, x2, 1.0f,
                                            out, nullptr, S_SEQ, DMODEL, FFDIM);
}
